// Decoder_26577257627794
// MI455X (gfx1250) — matrix operand order repaired and checked by reading, not compiled, not run on hardware
//
#include <hip/hip_runtime.h>
#include <hip/hip_bf16.h>

// ---- CDNA5 WMMA types ----
typedef __attribute__((ext_vector_type(16))) __bf16 v16bf;
typedef __attribute__((ext_vector_type(8)))  float  v8f;

union BF16Frag { v16bf v; uint4 q[2]; };

__device__ __forceinline__ unsigned short f2bf(float x) {
  unsigned int u = __float_as_uint(x);
  unsigned int lsb = (u >> 16) & 1u;
  return (unsigned short)((u + 0x7FFFu + lsb) >> 16);  // RNE
}

// Strip of NT 16x16 tiles sharing one A fragment per k-step, with explicit
// double buffering: fragment loads for k+32 are issued before the WMMAs of k,
// so the wave overlaps L2 latency with matrix math (low-occupancy recurrence).
// arow: A row (m0+l15), row-major bf16, K contiguous.
// brow: B^T row (ns+l15), tiles advance by 16*ldb rows.
// Layout per CDNA5 ISA 7.12.2 (16-bit A 16x32 / B 32x16):
//  A lane<16: K = k0+0..7 then k0+16..23 ; lane>=16: +8
//  B lane<16: K = k0+0..7 then k0+16..23 ; lane>=16: +8
template <int NT>
__device__ __forceinline__ void wmma_strip(const unsigned short* __restrict__ arow,
                                           const unsigned short* __restrict__ brow,
                                           long ldb, int K, int half, v8f* acc) {
  const int ao0 = (half ? 8 : 0);
  const int ao1 = 16 + ao0;
  const int bo  = (half ? 8 : 0);
  BF16Frag a, b[NT];
  a.q[0] = *(const uint4*)(arow + ao0);
  a.q[1] = *(const uint4*)(arow + ao1);
#pragma unroll
  for (int tn = 0; tn < NT; ++tn) {
    const uint4* pb = (const uint4*)(brow + (long)tn * 16 * ldb + bo);
    b[tn].q[0] = pb[0];
    b[tn].q[1] = pb[2];
  }
#pragma unroll
  for (int k0 = 0; k0 < K; k0 += 32) {
    const int kn = k0 + 32;
    const bool more = kn < K;
    BF16Frag an, bn[NT];
    if (more) {                                   // prefetch next k-step
      an.q[0] = *(const uint4*)(arow + kn + ao0);
      an.q[1] = *(const uint4*)(arow + kn + ao1);
#pragma unroll
      for (int tn = 0; tn < NT; ++tn) {
        const uint4* pb = (const uint4*)(brow + (long)tn * 16 * ldb + kn + bo);
        bn[tn].q[0] = pb[0];
        bn[tn].q[1] = pb[2];
      }
    }
#pragma unroll
    for (int tn = 0; tn < NT; ++tn)
      acc[tn] = __builtin_amdgcn_wmma_f32_16x16x32_bf16(false, a.v, false, b[tn].v,
                                                        (short)0, acc[tn], false, false);
    if (more) {
      a = an;
#pragma unroll
      for (int tn = 0; tn < NT; ++tn) b[tn] = bn[tn];
    }
  }
}

// ---------------- prep kernels ----------------

// input_token stream is data-independent of the float recurrence.
__global__ void k_prep_tokens(const int* __restrict__ tgt, int* __restrict__ toks) {
  int b = threadIdx.x;
  if (b >= 32) return;
  int fin = 0;
  toks[b] = 0;                       // step 0 input = sos (0)
  for (int t = 1; t < 191; ++t) {
    int nt = tgt[b * 192 + t];       // xs[t-1] = tgt[:, t]
    fin |= (nt == 130);              // EOS
    toks[t * 32 + b] = fin ? 0 : nt;
  }
}

__global__ void k_convert(const float* __restrict__ W_ih, const float* __restrict__ W_hh,
                          const float* __restrict__ W_v,  const float* __restrict__ W_c,
                          const float* __restrict__ W_out,const float* __restrict__ enc,
                          unsigned short* __restrict__ W_ihT, unsigned short* __restrict__ W_hhT,
                          unsigned short* __restrict__ W_vT,  unsigned short* __restrict__ W_cT,
                          unsigned short* __restrict__ W_outT,unsigned short* __restrict__ enc_bf) {
  const long n0 = 2048L * 768, n1 = 2048L * 512, n2 = 512L * 512,
             n3 = 512L * 1024, n4 = 640L * 512, n5 = 12288L * 512;
  const long total = n0 + n1 + n2 + n3 + n4 + n5;
  long stride = (long)gridDim.x * blockDim.x;
  for (long i = (long)blockIdx.x * blockDim.x + threadIdx.x; i < total; i += stride) {
    long j = i;
    if (j < n0) { long n = j / 768,  k = j % 768;  W_ihT[j] = f2bf(W_ih[k * 2048 + n]); continue; }
    j -= n0;
    if (j < n1) { long n = j / 512,  k = j % 512;  W_hhT[j] = f2bf(W_hh[k * 2048 + n]); continue; }
    j -= n1;
    if (j < n2) { long d = j / 512,  c = j % 512;  W_vT[j]  = f2bf(W_v [c * 512  + d]); continue; }
    j -= n2;
    if (j < n3) { long n = j / 1024, k = j % 1024; W_cT[j]  = f2bf(W_c [k * 512  + n]); continue; }
    j -= n3;
    if (j < n4) { long n = j / 512,  k = j % 512;
                  W_outT[j] = (n < 600) ? f2bf(W_out[k * 600 + n]) : (unsigned short)0; continue; }
    j -= n4;
    enc_bf[j] = f2bf(enc[j]);
  }
}

__global__ void k_emb(const float* __restrict__ table, const int* __restrict__ toks,
                      unsigned short* __restrict__ emb_seq) {
  const long total = 191L * 32 * 256;
  long stride = (long)gridDim.x * blockDim.x;
  for (long i = (long)blockIdx.x * blockDim.x + threadIdx.x; i < total; i += stride) {
    long e = i & 255, tb = i >> 8;              // tb = t*32+b
    int tok = toks[tb];
    emb_seq[i] = f2bf(table[(long)tok * 256 + e]);
  }
}

__global__ void k_init(float* __restrict__ hx, float* __restrict__ cx,
                       unsigned short* __restrict__ hx_bf,
                       unsigned short* __restrict__ A_buf,
                       const unsigned short* __restrict__ emb_seq) {
  int stride = gridDim.x * blockDim.x;
  for (int i = blockIdx.x * blockDim.x + threadIdx.x; i < 32 * 512; i += stride) {
    hx[i] = 0.f; cx[i] = 0.f; hx_bf[i] = 0;
  }
  for (int i = blockIdx.x * blockDim.x + threadIdx.x; i < 32 * 768; i += stride) {
    int b = i / 768, c2 = i % 768;              // [emb(256) | o_prev(512)=0]
    A_buf[i] = (c2 < 256) ? emb_seq[b * 256 + c2] : (unsigned short)0;
  }
}

// Wv = enc(12288x512) @ W_v(512x512), fp32 out (step-invariant precompute).
__global__ void k_wv(const unsigned short* __restrict__ enc_bf,
                     const unsigned short* __restrict__ W_vT,
                     float* __restrict__ Wv) {
  int wave = (blockIdx.x * 256 + threadIdx.x) >> 5;   // 0..6143 = 768 m-tiles x 8 strips
  int lane = threadIdx.x & 31;
  int m0 = (wave >> 3) * 16, ns = (wave & 7) * 64;
  int half = lane >> 4, l15 = lane & 15;
  v8f acc[4] = {};
  wmma_strip<4>(enc_bf + (long)(m0 + l15) * 512, W_vT + (long)(ns + l15) * 512,
                512, 512, half, acc);
  int rowbase = m0 + half * 8;
#pragma unroll
  for (int tn = 0; tn < 4; ++tn) {
    int n = ns + tn * 16 + l15;
#pragma unroll
    for (int r = 0; r < 8; ++r)
      Wv[(long)(rowbase + r) * 512 + n] = acc[tn][r];
  }
}

// ---------------- per-step kernels ----------------

// gates(32x2048) = [emb,o_prev](32x768)@W_ih + hx(32x512)@W_hh + b_ih + b_hh
__global__ void k_gates(const unsigned short* __restrict__ A_buf,
                        const unsigned short* __restrict__ hx_bf,
                        const unsigned short* __restrict__ W_ihT,
                        const unsigned short* __restrict__ W_hhT,
                        const float* __restrict__ b_ih, const float* __restrict__ b_hh,
                        float* __restrict__ gates) {
  int wave = (blockIdx.x * 256 + threadIdx.x) >> 5;   // 0..63 = 2 m-tiles x 32 strips
  int lane = threadIdx.x & 31;
  int m0 = (wave & 1) * 16, ns = (wave >> 1) * 64;
  int half = lane >> 4, l15 = lane & 15;
  v8f acc[4] = {};
  wmma_strip<4>(A_buf + (m0 + l15) * 768, W_ihT + (long)(ns + l15) * 768, 768, 768, half, acc);
  wmma_strip<4>(hx_bf + (m0 + l15) * 512, W_hhT + (long)(ns + l15) * 512, 512, 512, half, acc);
  int rowbase = m0 + half * 8;
#pragma unroll
  for (int tn = 0; tn < 4; ++tn) {
    int n = ns + tn * 16 + l15;
    float bias = b_ih[n] + b_hh[n];
#pragma unroll
    for (int r = 0; r < 8; ++r)
      gates[(rowbase + r) * 2048 + n] = acc[tn][r] + bias;
  }
}

__global__ void k_lstm(const float* __restrict__ gates, float* __restrict__ hx,
                       float* __restrict__ cx, unsigned short* __restrict__ hx_bf,
                       unsigned short* __restrict__ A2) {
  int i = blockIdx.x * blockDim.x + threadIdx.x;
  if (i >= 32 * 512) return;
  int b = i >> 9, j = i & 511;
  const float* g = gates + b * 2048;
  float ig = 1.f / (1.f + expf(-g[j]));
  float fg = 1.f / (1.f + expf(-g[512 + j]));
  float gg = tanhf(g[1024 + j]);
  float og = 1.f / (1.f + expf(-g[1536 + j]));
  float c = fg * cx[i] + ig * gg;
  float h = og * tanhf(c);
  cx[i] = c; hx[i] = h;
  unsigned short hb = f2bf(h);
  hx_bf[i] = hb;
  A2[b * 1024 + j] = hb;                      // [hx | ctx] staging for W_c GEMM
}

// Fused Bahdanau attention, one workgroup per batch row b.
__global__ void k_attn(const float* __restrict__ hx, const float* __restrict__ W_h,
                       const float* __restrict__ Wv, const float* __restrict__ vvec,
                       const float* __restrict__ enc, unsigned short* __restrict__ A2,
                       float* __restrict__ out_alpha, int t) {
  __shared__ float hxs[512];
  __shared__ float q[512];
  __shared__ float sc[384];
  __shared__ float red[256];
  int b = blockIdx.x, tid = threadIdx.x;
  for (int d = tid; d < 512; d += 256) hxs[d] = hx[b * 512 + d];
  __syncthreads();
  for (int d = tid; d < 512; d += 256) {       // q = hx @ W_h
    float s = 0.f;
    for (int k = 0; k < 512; ++k) s += hxs[k] * W_h[k * 512 + d];
    q[d] = s;
  }
  __syncthreads();
  int wave = tid >> 5, lane = tid & 31;
  for (int n = wave; n < 384; n += 8) {        // score[n] = v . tanh(q + Wv[b,n])
    const float* wp = Wv + ((long)b * 384 + n) * 512;
    float s = 0.f;
    for (int d = lane; d < 512; d += 32) s += vvec[d] * tanhf(q[d] + wp[d]);
    for (int off = 16; off > 0; off >>= 1) s += __shfl_xor(s, off, 32);
    if (lane == 0) sc[n] = s;
  }
  __syncthreads();
  float m = -3.4e38f;
  for (int n = tid; n < 384; n += 256) m = fmaxf(m, sc[n]);
  red[tid] = m; __syncthreads();
  for (int s2 = 128; s2 > 0; s2 >>= 1) { if (tid < s2) red[tid] = fmaxf(red[tid], red[tid + s2]); __syncthreads(); }
  float mx = red[0]; __syncthreads();
  float ps = 0.f;
  for (int n = tid; n < 384; n += 256) { float e = expf(sc[n] - mx); sc[n] = e; ps += e; }
  red[tid] = ps; __syncthreads();
  for (int s2 = 128; s2 > 0; s2 >>= 1) { if (tid < s2) red[tid] += red[tid + s2]; __syncthreads(); }
  float inv = 1.f / red[0]; __syncthreads();
  for (int n = tid; n < 384; n += 256) {
    float a = sc[n] * inv;
    sc[n] = a;
    out_alpha[((long)b * 191 + t) * 384 + n] = a;
  }
  __syncthreads();
  for (int c = tid; c < 512; c += 256) {       // context = alpha @ enc[b]
    float s = 0.f;
    const float* ep = enc + (long)b * 384 * 512 + c;
    for (int n = 0; n < 384; ++n) s += sc[n] * ep[n * 512];
    A2[b * 1024 + 512 + c] = f2bf(s);
  }
}

// o_t = tanh([hx,ctx](32x1024) @ W_c + b_c); also stage next-step A rows.
__global__ void k_out1(const unsigned short* __restrict__ A2,
                       const unsigned short* __restrict__ W_cT,
                       const float* __restrict__ b_c,
                       unsigned short* __restrict__ o_bf,
                       unsigned short* __restrict__ A_buf,
                       const unsigned short* __restrict__ emb_seq, int t) {
  if (blockIdx.x == 2) {                       // stage emb for step t+1
    if (t + 1 < 191) {
      for (int i = threadIdx.x; i < 32 * 256; i += 256) {
        int b = i >> 8, e = i & 255;
        A_buf[b * 768 + e] = emb_seq[((long)(t + 1) * 32 + b) * 256 + e];
      }
    }
    return;
  }
  int wave = (blockIdx.x * 256 + threadIdx.x) >> 5;  // 0..15 = 2 m-tiles x 8 strips
  int lane = threadIdx.x & 31;
  int m0 = (wave & 1) * 16, ns = (wave >> 1) * 64;
  int half = lane >> 4, l15 = lane & 15;
  v8f acc[4] = {};
  wmma_strip<4>(A2 + (m0 + l15) * 1024, W_cT + (long)(ns + l15) * 1024, 1024, 1024, half, acc);
  int rowbase = m0 + half * 8;
#pragma unroll
  for (int tn = 0; tn < 4; ++tn) {
    int n = ns + tn * 16 + l15;
    float bias = b_c[n];
#pragma unroll
    for (int r = 0; r < 8; ++r) {
      float o = tanhf(acc[tn][r] + bias);
      unsigned short ob = f2bf(o);
      int row = rowbase + r;
      o_bf[row * 512 + n] = ob;
      A_buf[row * 768 + 256 + n] = ob;         // o_prev for next step
    }
  }
}

// logits = o_t(32x512) @ W_out(512x600, padded to 640) + b_out
__global__ void k_out2(const unsigned short* __restrict__ o_bf,
                       const unsigned short* __restrict__ W_outT,
                       const float* __restrict__ b_out,
                       float* __restrict__ out_logits, int t) {
  int wave = (blockIdx.x * 256 + threadIdx.x) >> 5;
  if (wave >= 20) return;                      // whole-wave exit (EXEC all-1s for WMMA)
  int lane = threadIdx.x & 31;
  int m0 = (wave & 1) * 16, ns = (wave >> 1) * 64;
  int half = lane >> 4, l15 = lane & 15;
  v8f acc[4] = {};
  wmma_strip<4>(o_bf + (m0 + l15) * 512, W_outT + (long)(ns + l15) * 512, 512, 512, half, acc);
  int rowbase = m0 + half * 8;
#pragma unroll
  for (int tn = 0; tn < 4; ++tn) {
    int n = ns + tn * 16 + l15;
    if (n >= 600) continue;                    // skip pad columns (after WMMA)
    float bias = b_out[n];
#pragma unroll
    for (int r = 0; r < 8; ++r) {
      int b = rowbase + r;
      out_logits[((long)b * 191 + t) * 600 + n] = acc[tn][r] + bias;
    }
  }
}

// ---------------- host launcher ----------------

extern "C" void kernel_launch(void* const* d_in, const int* in_sizes, int n_in,
                              void* d_out, int out_size, void* d_ws, size_t ws_size,
                              hipStream_t stream) {
  (void)in_sizes; (void)n_in; (void)out_size; (void)ws_size;
  const float* enc       = (const float*)d_in[0];
  const int*   tgt       = (const int*)  d_in[1];
  const float* emb_table = (const float*)d_in[2];
  const float* W_ih      = (const float*)d_in[3];
  const float* b_ih      = (const float*)d_in[4];
  const float* W_hh      = (const float*)d_in[5];
  const float* b_hh      = (const float*)d_in[6];
  const float* W_h       = (const float*)d_in[7];
  const float* W_v       = (const float*)d_in[8];
  const float* vvec      = (const float*)d_in[9];
  const float* W_c       = (const float*)d_in[10];
  const float* b_c       = (const float*)d_in[11];
  const float* W_out     = (const float*)d_in[12];
  const float* b_out     = (const float*)d_in[13];

  float* out_logits = (float*)d_out;
  float* out_alpha  = out_logits + (size_t)32 * 191 * 600;

  // workspace carve-out (~49 MB), 256B aligned
  char* w = (char*)d_ws;
  auto carve = [&](size_t bytes) { char* p = w; w += (bytes + 255) & ~(size_t)255; return p; };
  unsigned short* W_ihT  = (unsigned short*)carve(2048ull * 768 * 2);
  unsigned short* W_hhT  = (unsigned short*)carve(2048ull * 512 * 2);
  unsigned short* W_vT   = (unsigned short*)carve(512ull * 512 * 2);
  unsigned short* W_cT   = (unsigned short*)carve(512ull * 1024 * 2);
  unsigned short* W_outT = (unsigned short*)carve(640ull * 512 * 2);
  unsigned short* enc_bf = (unsigned short*)carve(12288ull * 512 * 2);
  unsigned short* emb_seq= (unsigned short*)carve(191ull * 32 * 256 * 2);
  int*            toks   = (int*)           carve(191ull * 32 * 4);
  float*          Wv     = (float*)         carve(12288ull * 512 * 4);
  float*          hx     = (float*)         carve(32ull * 512 * 4);
  float*          cx     = (float*)         carve(32ull * 512 * 4);
  unsigned short* hx_bf  = (unsigned short*)carve(32ull * 512 * 2);
  unsigned short* o_bf   = (unsigned short*)carve(32ull * 512 * 2);
  unsigned short* A_buf  = (unsigned short*)carve(32ull * 768 * 2);   // [emb | o_prev]
  unsigned short* A2     = (unsigned short*)carve(32ull * 1024 * 2);  // [hx | ctx]
  float*          gates  = (float*)         carve(32ull * 2048 * 4);

  k_prep_tokens<<<1, 32, 0, stream>>>(tgt, toks);
  k_convert<<<4096, 256, 0, stream>>>(W_ih, W_hh, W_v, W_c, W_out, enc,
                                      W_ihT, W_hhT, W_vT, W_cT, W_outT, enc_bf);
  k_emb<<<2048, 256, 0, stream>>>(emb_table, toks, emb_seq);
  k_init<<<256, 256, 0, stream>>>(hx, cx, hx_bf, A_buf, emb_seq);
  k_wv<<<768, 256, 0, stream>>>(enc_bf, W_vT, Wv);

  for (int t = 0; t < 191; ++t) {
    k_gates<<<8, 256, 0, stream>>>(A_buf, hx_bf, W_ihT, W_hhT, b_ih, b_hh, gates);
    k_lstm<<<64, 256, 0, stream>>>(gates, hx, cx, hx_bf, A2);
    k_attn<<<32, 256, 0, stream>>>(hx, W_h, Wv, vvec, enc, A2, out_alpha, t);
    k_out1<<<3, 256, 0, stream>>>(A2, W_cT, b_c, o_bf, A_buf, emb_seq, t);
    k_out2<<<3, 256, 0, stream>>>(o_bf, W_outT, b_out, out_logits, t);
  }
}
